// VariationalGCNEncoder_60756607369296
// MI455X (gfx1250) — matrix operand order repaired and checked by reading, not compiled, not run on hardware
//
#include <hip/hip_runtime.h>
#include <hip/hip_bf16.h>

// ---------------------------------------------------------------------------
// Variational GCN encoder for MI455X (gfx1250, wave32, WMMA).
//   h1 = ReLU(BN(Agg(x @ W1)))          K=256 -> 128
//   h2 = ReLU(BN(Agg(h1 @ W2)))         K=128 -> 128
//   [mu|ls] = Agg(h2 @ [Wmu|Wls]) + b   K=128 -> 128 (fused heads)
// Agg = symmetric-normalized scatter-add over edges + self loops.
// Hot buffers (~110MB) are L2-resident (192MB) -> edge atomics are L2-local.
// GEMMs: v_wmma_f32_16x16x32_bf16, operands prepacked in contiguous-k fragment order
// so the inner loop is 1x b64 stage + 2x ds_load_b128 + 2x global_load_b128
// + 1 wmma per 8192 MACs.
// ---------------------------------------------------------------------------

#define Nn   100000
#define Ee   1600000
#define INF_ 256
#define HID_ 128
#define OUT_ 64
#define EPSF 1e-5f

typedef __attribute__((ext_vector_type(16))) __bf16 v16bf;
typedef __attribute__((ext_vector_type(8)))  float  v8f;

static __device__ __forceinline__ void atomic_add_f32(float* p, float v) {
  __hip_atomic_fetch_add(p, v, __ATOMIC_RELAXED, __HIP_MEMORY_SCOPE_AGENT);
}

// ---------------- graph normalization ----------------
__global__ void k_init_deg(float* deg) {
  int i = blockIdx.x * blockDim.x + threadIdx.x;
  if (i < Nn) deg[i] = 1.0f;                 // self loop
}
__global__ void k_deg_edges(const int* __restrict__ dst, float* __restrict__ deg) {
  int e = blockIdx.x * blockDim.x + threadIdx.x;
  if (e < Ee) atomic_add_f32(&deg[dst[e]], 1.0f);
}
__global__ void k_dinv(const float* __restrict__ deg, float* __restrict__ dinv) {
  int i = blockIdx.x * blockDim.x + threadIdx.x;
  if (i < Nn) dinv[i] = rsqrtf(deg[i]);      // deg >= 1 always
}
__global__ void k_norm(const int* __restrict__ src, const int* __restrict__ dst,
                       const float* __restrict__ dinv, float* __restrict__ norm) {
  int e = blockIdx.x * blockDim.x + threadIdx.x;
  if (e < Ee) norm[e] = dinv[src[e]] * dinv[dst[e]];
}

// ---------------- weight prepack: f32 row-major -> bf16 B-fragment order ---
// Layout: P[((kstep*8 + coltile)*32 + lane)*16 + j], where for B(32x16) in contiguous-k order:
//   lane = coltile-local column + 16*kHalf-group;  k = kstep*32 + (lane>>4)*16 + j
__global__ void k_pack_w(const float* __restrict__ W, __bf16* __restrict__ P, int K) {
  int idx = blockIdx.x * blockDim.x + threadIdx.x;   // K*128 elements
  if (idx >= K * 128) return;
  int j     = idx & 15;
  int lane  = (idx >> 4) & 31;
  int ct    = (idx >> 9) & 7;
  int kstep = idx >> 12;
  int k   = (kstep << 5) + ((lane >> 4) << 4) + j;
  int col = (ct << 4) + (lane & 15);
  P[idx] = (__bf16)W[k * 128 + col];
}

// ---------------- WMMA GEMM: C[N,128] = A[N,K] * W[K,128] ----------------
// blockDim = 256 (8 waves). One 16-row tile per block, one 16-col tile per
// wave. A tile staged in LDS as bf16 in A-fragment lane order.
__global__ __launch_bounds__(256)
void k_gemm_wmma(const float* __restrict__ A, const __bf16* __restrict__ P,
                 float* __restrict__ C, int K) {
  __shared__ __align__(32) __bf16 lds_a[512];   // 16x32 tile, fragment order
  const int tid  = threadIdx.x;
  const int wave = tid >> 5;                    // 0..7 -> column tile
  const int lane = tid & 31;
  const int half = lane >> 4;
  const int m16  = lane & 15;
  const int rowBase = blockIdx.x * 16;

  // A staging: thread -> (row sm, even-k pair k0); swizzle to A(16x32):
  //   lanes 0-15 hold K {0..15}, lanes 16-31 hold K {16..31};
  //   within-lane j = k & 15
  const int sm    = tid >> 4;                   // 0..15
  const int k0    = (tid & 15) << 1;            // 0,2,..,30
  const int slane = sm + ((k0 >> 4) << 4);
  const int sj    = k0 & 15;
  const float* arow = A + (size_t)(rowBase + sm) * K + k0;

  v8f acc = {};
  const int kSteps = K >> 5;                    // K = 256 or 128
  for (int ks = 0; ks < kSteps; ++ks) {
    float2 f = *(const float2*)(arow + ((size_t)ks << 5));
    lds_a[slane * 16 + sj]     = (__bf16)f.x;
    lds_a[slane * 16 + sj + 1] = (__bf16)f.y;
    __syncthreads();
    v16bf a = *(const v16bf*)(lds_a + lane * 16);                      // 32B LDS
    v16bf b = *(const v16bf*)(P + ((size_t)(((ks << 3) + wave) << 5 | lane) << 4));
    acc = __builtin_amdgcn_wmma_f32_16x16x32_bf16(
        /*neg_a=*/false, a, /*neg_b=*/false, b,
        /*c_mod=*/(short)0, acc, /*reuse_a=*/false, /*reuse_b=*/false);
    __syncthreads();
  }
  // C/D layout: VGPR r -> row = 8*half + r, lane&15 -> col
#pragma unroll
  for (int r = 0; r < 8; ++r)
    C[(size_t)(rowBase + 8 * half + r) * 128 + wave * 16 + m16] = acc[r];
}

// ---------------- scatter-add (edges): one wave per edge, float4/lane ------
__global__ void k_zero(float* p, long long n) {
  long long i = (long long)blockIdx.x * blockDim.x + threadIdx.x;
  if (i < n) p[i] = 0.0f;
}
__global__ void k_scatter_edges(const float* __restrict__ h,
                                const int* __restrict__ src,
                                const int* __restrict__ dst,
                                const float* __restrict__ norm,
                                float* __restrict__ agg) {
  long long gid = (long long)blockIdx.x * blockDim.x + threadIdx.x;
  long long e = gid >> 5;
  if (e >= Ee) return;
  int lane = (int)(gid & 31);
  int s = src[e], d = dst[e];
  float w = norm[e];
  const float4 v = *(const float4*)(h + (size_t)s * 128 + lane * 4);
  float* o = agg + (size_t)d * 128 + lane * 4;
  atomic_add_f32(o + 0, v.x * w);
  atomic_add_f32(o + 1, v.y * w);
  atomic_add_f32(o + 2, v.z * w);
  atomic_add_f32(o + 3, v.w * w);
}
// self loops: unique row per wave -> plain RMW (runs after edge kernel)
__global__ void k_self_loops(const float* __restrict__ h,
                             const float* __restrict__ dinv,
                             float* __restrict__ agg) {
  long long gid = (long long)blockIdx.x * blockDim.x + threadIdx.x;
  long long i = gid >> 5;
  if (i >= Nn) return;
  int lane = (int)(gid & 31);
  float w = dinv[i] * dinv[i];
  const float4 v = *(const float4*)(h + (size_t)i * 128 + lane * 4);
  float4* o = (float4*)(agg + (size_t)i * 128 + lane * 4);
  float4 x = *o;
  x.x += v.x * w; x.y += v.y * w; x.z += v.z * w; x.w += v.w * w;
  *o = x;
}

// ---------------- BatchNorm (+ReLU) ----------------
__global__ void k_bn_stats(const float* __restrict__ h, float* __restrict__ sum,
                           float* __restrict__ sumsq) {
  const int col = threadIdx.x;               // blockDim = 128
  long long row0 = (long long)blockIdx.x * 128;
  float s = 0.f, q = 0.f;
  for (int r = 0; r < 128; ++r) {
    long long row = row0 + r;
    if (row < Nn) {
      float v = h[row * 128 + col];
      s += v; q += v * v;
    }
  }
  atomic_add_f32(&sum[col], s);
  atomic_add_f32(&sumsq[col], q);
}
__global__ void k_bn_final(const float* __restrict__ sum, const float* __restrict__ sumsq,
                           const float* __restrict__ g, const float* __restrict__ bt,
                           float* __restrict__ scale, float* __restrict__ shift) {
  int c = threadIdx.x;                       // 128 threads
  float mean = sum[c] * (1.0f / Nn);
  float var  = sumsq[c] * (1.0f / Nn) - mean * mean;
  float sc   = g[c] * rsqrtf(var + EPSF);
  scale[c] = sc;
  shift[c] = bt[c] - mean * sc;              // conv bias cancels inside BN
}
__global__ void k_bn_apply_relu(float* __restrict__ h, const float* __restrict__ scale,
                                const float* __restrict__ shift) {
  long long i = (long long)blockIdx.x * blockDim.x + threadIdx.x;
  if (i >= (long long)Nn * 128) return;
  int c = (int)(i & 127);
  float v = h[i] * scale[c] + shift[c];
  h[i] = fmaxf(v, 0.0f);
}

// ---------------- fused output head ----------------
__global__ void k_fuse_w3(const float* __restrict__ Wmu, const float* __restrict__ Wls,
                          float* __restrict__ W3) {
  int i = blockIdx.x * blockDim.x + threadIdx.x;   // 128*128
  if (i >= 128 * 128) return;
  int k = i >> 7, j = i & 127;
  W3[i] = (j < 64) ? Wmu[k * 64 + j] : Wls[k * 64 + (j - 64)];
}
__global__ void k_write_out(const float* __restrict__ agg,
                            const float* __restrict__ bmu, const float* __restrict__ bls,
                            float* __restrict__ out) {
  long long i = (long long)blockIdx.x * blockDim.x + threadIdx.x;
  if (i >= (long long)Nn * 128) return;
  long long node = i >> 7;
  int j = (int)(i & 127);
  if (j < 64)
    out[node * 64 + j] = agg[i] + bmu[j];                                  // mu
  else
    out[(long long)Nn * 64 + node * 64 + (j - 64)] = agg[i] + bls[j - 64]; // logstd
}

// ---------------------------------------------------------------------------
extern "C" void kernel_launch(void* const* d_in, const int* in_sizes, int n_in,
                              void* d_out, int out_size, void* d_ws, size_t ws_size,
                              hipStream_t stream) {
  const float* x   = (const float*)d_in[0];
  const float* W1  = (const float*)d_in[1];
  const float* g1  = (const float*)d_in[3];
  const float* bt1 = (const float*)d_in[4];
  const float* W2  = (const float*)d_in[5];
  const float* g2  = (const float*)d_in[7];
  const float* bt2 = (const float*)d_in[8];
  const float* Wmu = (const float*)d_in[9];
  const float* bmu = (const float*)d_in[10];
  const float* Wls = (const float*)d_in[11];
  const float* bls = (const float*)d_in[12];
  const int*   src = (const int*)d_in[13];
  const int*   dst = src + Ee;
  float* out = (float*)d_out;

  // workspace layout (floats)
  float* ws    = (float*)d_ws;
  const long long NH = (long long)Nn * 128;
  float* bufA  = ws;                 // GEMM output h
  float* bufB  = bufA + NH;          // aggregation buffer
  float* norm  = bufB + NH;          // [E]
  float* deg   = norm + Ee;          // [N]
  float* dinv  = deg + Nn;           // [N]
  float* W3    = dinv + Nn;          // [128*128] fused f32 head weights
  float* sum   = W3 + 128 * 128;     // [128]
  float* sumsq = sum + 128;          // [128]
  float* scale = sumsq + 128;        // [128]
  float* shift = scale + 128;        // [128]
  __bf16* P1 = (__bf16*)(shift + 128);        // [256*128] packed W1
  __bf16* P2 = P1 + 256 * 128;                // [128*128] packed W2
  __bf16* P3 = P2 + 128 * 128;                // [128*128] packed W3

  const int T = 256;
  const int gN   = (Nn + T - 1) / T;
  const int gE   = (Ee + T - 1) / T;
  const int gE32 = (int)(((long long)Ee * 32 + T - 1) / T);
  const int gN32 = (int)(((long long)Nn * 32 + T - 1) / T);
  const int gNH  = (int)((NH + T - 1) / T);
  const int gBN  = (Nn + 127) / 128;
  const int gGEMM = Nn / 16;         // 6250, exact

  // --- graph normalization ---
  k_init_deg<<<gN, T, 0, stream>>>(deg);
  k_deg_edges<<<gE, T, 0, stream>>>(dst, deg);
  k_dinv<<<gN, T, 0, stream>>>(deg, dinv);
  k_norm<<<gE, T, 0, stream>>>(src, dst, dinv, norm);

  // --- weight prepack (runs once per call; tiny) ---
  k_pack_w<<<(256 * 128) / T, T, 0, stream>>>(W1, P1, INF_);
  k_pack_w<<<(128 * 128) / T, T, 0, stream>>>(W2, P2, HID_);
  k_fuse_w3<<<(128 * 128) / T, T, 0, stream>>>(Wmu, Wls, W3);
  k_pack_w<<<(128 * 128) / T, T, 0, stream>>>(W3, P3, HID_);

  // --- layer 1: conv(x, W1) -> BN -> ReLU ---
  k_gemm_wmma<<<gGEMM, 256, 0, stream>>>(x, P1, bufA, INF_);
  k_zero<<<gNH, T, 0, stream>>>(bufB, NH);
  k_scatter_edges<<<gE32, T, 0, stream>>>(bufA, src, dst, norm, bufB);
  k_self_loops<<<gN32, T, 0, stream>>>(bufA, dinv, bufB);
  k_zero<<<1, 256, 0, stream>>>(sum, 256);   // sum+sumsq contiguous
  k_bn_stats<<<gBN, 128, 0, stream>>>(bufB, sum, sumsq);
  k_bn_final<<<1, 128, 0, stream>>>(sum, sumsq, g1, bt1, scale, shift);
  k_bn_apply_relu<<<gNH, T, 0, stream>>>(bufB, scale, shift);

  // --- layer 2: conv(h1, W2) -> BN -> ReLU ---
  k_gemm_wmma<<<gGEMM, 256, 0, stream>>>(bufB, P2, bufA, HID_);
  k_zero<<<gNH, T, 0, stream>>>(bufB, NH);
  k_scatter_edges<<<gE32, T, 0, stream>>>(bufA, src, dst, norm, bufB);
  k_self_loops<<<gN32, T, 0, stream>>>(bufA, dinv, bufB);
  k_zero<<<1, 256, 0, stream>>>(sum, 256);
  k_bn_stats<<<gBN, 128, 0, stream>>>(bufB, sum, sumsq);
  k_bn_final<<<1, 128, 0, stream>>>(sum, sumsq, g2, bt2, scale, shift);
  k_bn_apply_relu<<<gNH, T, 0, stream>>>(bufB, scale, shift);

  // --- fused heads: conv(h2, [Wmu|Wls]) + bias -> (mu, logstd) ---
  k_gemm_wmma<<<gGEMM, 256, 0, stream>>>(bufB, P3, bufA, HID_);
  k_zero<<<gNH, T, 0, stream>>>(bufB, NH);
  k_scatter_edges<<<gE32, T, 0, stream>>>(bufA, src, dst, norm, bufB);
  k_self_loops<<<gN32, T, 0, stream>>>(bufA, dinv, bufB);
  k_write_out<<<gNH, T, 0, stream>>>(bufB, bmu, bls, out);
}
